// DeepGeneRegulationNetwork_32839319945348
// MI455X (gfx1250) — matrix operand order repaired and checked by reading, not compiled, not run on hardware
//
#include <hip/hip_runtime.h>
#include <hip/hip_bf16.h>
#include <math.h>

// ---------------------------------------------------------------------------
// CDNA5 (gfx1250) implementation. wave32. All dense algebra goes through
// v_wmma_f32_16x16x32_f16 (f16 inputs, f32 accumulate).
// ---------------------------------------------------------------------------

typedef __attribute__((ext_vector_type(16))) _Float16 v16h;
typedef __attribute__((ext_vector_type(8)))  _Float16 v8h;
typedef __attribute__((ext_vector_type(4)))  _Float16 v4h;
typedef __attribute__((ext_vector_type(8)))  float    v8f;
typedef __attribute__((ext_vector_type(4)))  float    v4f;

#define NN   2048
#define HH   256
#define NHD  8
#define HDD  32

#define BM   64
#define BN   64
#define BK   32
#define LDP  48   // LDS pitch in halves (96 B -> keeps b128 LDS loads 16B aligned)

// ---------------------------------------------------------------------------
// WMMA GEMM: C = alpha * (A @ B) [+ bias] [+ C_in], optional act.
//   A: MxK row-major (lda).  REQUIREMENTS (all call sites honor them):
//      M % 64 == 0, K % 32 == 0, lda % 4 == 0, A 16B-aligned.
//   B: KxN row-major (ldb), or if transB: B[k][n] = Bsrc[n*ldb+k]
//      (transB callers additionally have N % 64 == 0, ldb % 4 == 0).
//   act: 0=none, 1=relu, 2=sigmoid
// Block = 256 threads = 8 waves. Block tile 64x64, K step 32.
// Wave w: rows [ (w>>2)*32, +32 ) as two 16x16 m-tiles, cols [ (w&3)*16, +16 ).
// ---------------------------------------------------------------------------
__global__ __launch_bounds__(256) void k_wmma_gemm(
    const float* __restrict__ A, int lda,
    const float* __restrict__ B, int ldb, int transB,
    float* __restrict__ C, int ldc,
    const float* __restrict__ bias,
    int M, int N, int K, float alpha, int act, int addC)
{
    __shared__ _Float16 Ah[BM][LDP];   // row-major, K minor
    __shared__ _Float16 Bt[BN][LDP];   // n-major, K minor (transposed tile)

    const int tid  = threadIdx.x;
    const int wave = tid >> 5;
    const int lane = tid & 31;
    const int mw   = wave >> 2;        // 0..1
    const int nw   = wave & 3;         // 0..3
    const int m0   = blockIdx.y * BM;
    const int n0   = blockIdx.x * BN;
    const int l16  = lane & 15;
    const int half = lane >> 4;

    v8f acc0 = {}, acc1 = {};

    for (int k0 = 0; k0 < K; k0 += BK) {
        // ---- stage A (BM x BK): unconditional b128 loads, ds_store_b64 ----
        #pragma unroll
        for (int j = 0; j < 2; ++j) {
            int idx = tid + j * 256;
            int r = idx >> 3, cv = (idx & 7) * 4;
            v4f a = *(const v4f*)(A + (size_t)(m0 + r) * lda + k0 + cv);
            v4h h4;
            h4[0] = (_Float16)a[0]; h4[1] = (_Float16)a[1];
            h4[2] = (_Float16)a[2]; h4[3] = (_Float16)a[3];
            *(v4h*)&Ah[r][cv] = h4;
        }
        // ---- stage B (BK x BN) transposed into Bt[n][k] ----
        if (transB) {
            // B[n][k] contiguous in k: b128 loads (N%64==0 for these callers)
            #pragma unroll
            for (int j = 0; j < 2; ++j) {
                int idx = tid + j * 256;
                int nn = idx >> 3, kv = (idx & 7) * 4;
                v4f b = *(const v4f*)(B + (size_t)(n0 + nn) * ldb + k0 + kv);
                v4h h4;
                h4[0] = (_Float16)b[0]; h4[1] = (_Float16)b[1];
                h4[2] = (_Float16)b[2]; h4[3] = (_Float16)b[3];
                *(v4h*)&Bt[nn][kv] = h4;
            }
        } else {
            // B[k][n]: 4 coalesced b32 loads per thread (fixed n, k..k+3),
            // branchless tail handling via clamped index + valid multiplier.
            #pragma unroll
            for (int j = 0; j < 2; ++j) {
                int idx = tid + j * 256;
                int kq = (idx >> 6) * 4, nn = idx & 63;
                int gn  = n0 + nn;
                int gns = gn < N ? gn : 0;
                float s = gn < N ? 1.0f : 0.0f;
                v4h h4;
                #pragma unroll
                for (int u = 0; u < 4; ++u)
                    h4[u] = (_Float16)(s * B[(size_t)(k0 + kq + u) * ldb + gns]);
                *(v4h*)&Bt[nn][kq] = h4;
            }
        }
        __syncthreads();

        // speculative prefetch of next A tile (global_prefetch_b8)
        if (k0 + BK < K)
            __builtin_prefetch(A + (size_t)(m0 + (tid >> 2)) * lda + k0 + BK, 0, 1);

        // ---- fragments: two contiguous ds_load_b128 runs each ----
        // B 32x16: vgprs 0-3 hold K = half*8 .. +7 ; vgprs 4-7 hold K = 16+half*8 .. +7
        const _Float16* bp = &Bt[nw * 16 + l16][half * 8];
        v8h b_lo = *(const v8h*)(bp);
        v8h b_hi = *(const v8h*)(bp + 16);
        v16h bf = __builtin_shufflevector(b_lo, b_hi,
                    0,1,2,3,4,5,6,7,8,9,10,11,12,13,14,15);

        // A 16x32: vgprs 0-3 hold K = half*8 .. +7 ; vgprs 4-7 hold K = 16+half*8 .. +7
        const int r0 = mw * 32 + l16;
        v8h a0lo = *(const v8h*)&Ah[r0][half * 8];
        v8h a0hi = *(const v8h*)&Ah[r0][16 + half * 8];
        v16h af0 = __builtin_shufflevector(a0lo, a0hi,
                    0,1,2,3,4,5,6,7,8,9,10,11,12,13,14,15);
        v8h a1lo = *(const v8h*)&Ah[r0 + 16][half * 8];
        v8h a1hi = *(const v8h*)&Ah[r0 + 16][16 + half * 8];
        v16h af1 = __builtin_shufflevector(a1lo, a1hi,
                    0,1,2,3,4,5,6,7,8,9,10,11,12,13,14,15);

        acc0 = __builtin_amdgcn_wmma_f32_16x16x32_f16(
                   false, af0, false, bf, (short)0, acc0, false, false);
        acc1 = __builtin_amdgcn_wmma_f32_16x16x32_f16(
                   false, af1, false, bf, (short)0, acc1, false, false);
        __syncthreads();
    }

    // ---- epilogue: C layout vgpr r -> row r + 8*half, col = lane&15 ----
    const int col = n0 + nw * 16 + l16;
    if (col < N) {
        #pragma unroll
        for (int r = 0; r < 8; ++r) {
            int row = m0 + mw * 32 + r + 8 * half;
            float v = alpha * acc0[r];
            if (bias) v += bias[col];
            size_t off = (size_t)row * ldc + col;
            if (addC) v += C[off];
            if (act == 1)      v = fmaxf(v, 0.0f);
            else if (act == 2) v = 1.0f / (1.0f + expf(-v));
            C[off] = v;
        }
        #pragma unroll
        for (int r = 0; r < 8; ++r) {
            int row = m0 + mw * 32 + 16 + r + 8 * half;
            float v = alpha * acc1[r];
            if (bias) v += bias[col];
            size_t off = (size_t)row * ldc + col;
            if (addC) v += C[off];
            if (act == 1)      v = fmaxf(v, 0.0f);
            else if (act == 2) v = 1.0f / (1.0f + expf(-v));
            C[off] = v;
        }
    }
}

// ---------------------------------------------------------------------------
// LayerNorm over rows of length D (population variance, eps=1e-5).
// act: 0=none, 1=relu, 2=gelu(exact)
// ---------------------------------------------------------------------------
__global__ __launch_bounds__(256) void k_layernorm(
    const float* __restrict__ X, int ldx, float* __restrict__ Y, int ldy,
    const float* __restrict__ gamma, const float* __restrict__ beta,
    int D, int act)
{
    __shared__ float sh1[256], sh2[256];
    const int row = blockIdx.x;
    const float* xr = X + (size_t)row * ldx;
    float s = 0.0f, s2 = 0.0f;
    for (int i = threadIdx.x; i < D; i += 256) {
        float v = xr[i]; s += v; s2 += v * v;
    }
    sh1[threadIdx.x] = s; sh2[threadIdx.x] = s2;
    __syncthreads();
    for (int off = 128; off > 0; off >>= 1) {
        if (threadIdx.x < off) {
            sh1[threadIdx.x] += sh1[threadIdx.x + off];
            sh2[threadIdx.x] += sh2[threadIdx.x + off];
        }
        __syncthreads();
    }
    const float mean = sh1[0] / (float)D;
    const float var  = sh2[0] / (float)D - mean * mean;
    const float inv  = rsqrtf(var + 1e-5f);
    for (int i = threadIdx.x; i < D; i += 256) {
        float v = (xr[i] - mean) * inv * gamma[i] + beta[i];
        if (act == 1)      v = fmaxf(v, 0.0f);
        else if (act == 2) v = 0.5f * v * (1.0f + erff(v * 0.70710678118654752f));
        Y[(size_t)row * ldy + i] = v;
    }
}

// ---------------------------------------------------------------------------
// Row softmax (row length Ncols), in place.
// ---------------------------------------------------------------------------
__global__ __launch_bounds__(256) void k_row_softmax(float* __restrict__ S, int Ncols)
{
    __shared__ float sh[256];
    const int row = blockIdx.x;
    float* r = S + (size_t)row * Ncols;
    float m = -INFINITY;
    for (int i = threadIdx.x; i < Ncols; i += 256) m = fmaxf(m, r[i]);
    sh[threadIdx.x] = m; __syncthreads();
    for (int off = 128; off > 0; off >>= 1) {
        if (threadIdx.x < off) sh[threadIdx.x] = fmaxf(sh[threadIdx.x], sh[threadIdx.x + off]);
        __syncthreads();
    }
    m = sh[0]; __syncthreads();
    float s = 0.0f;
    for (int i = threadIdx.x; i < Ncols; i += 256) {
        float e = expf(r[i] - m); r[i] = e; s += e;
    }
    sh[threadIdx.x] = s; __syncthreads();
    for (int off = 128; off > 0; off >>= 1) {
        if (threadIdx.x < off) sh[threadIdx.x] += sh[threadIdx.x + off];
        __syncthreads();
    }
    const float inv = 1.0f / sh[0];
    for (int i = threadIdx.x; i < Ncols; i += 256) r[i] *= inv;
}

// small-vector softmax (n <= 256), single block
__global__ __launch_bounds__(256) void k_softmax_vec(
    const float* __restrict__ p, int n, float* __restrict__ out)
{
    __shared__ float sh[256];
    float m = -INFINITY;
    for (int i = threadIdx.x; i < n; i += 256) m = fmaxf(m, p[i]);
    sh[threadIdx.x] = m; __syncthreads();
    for (int off = 128; off > 0; off >>= 1) {
        if (threadIdx.x < off) sh[threadIdx.x] = fmaxf(sh[threadIdx.x], sh[threadIdx.x + off]);
        __syncthreads();
    }
    m = sh[0]; __syncthreads();
    float s = 0.0f;
    for (int i = threadIdx.x; i < n; i += 256) s += expf(p[i] - m);
    sh[threadIdx.x] = s; __syncthreads();
    for (int off = 128; off > 0; off >>= 1) {
        if (threadIdx.x < off) sh[threadIdx.x] += sh[threadIdx.x + off];
        __syncthreads();
    }
    const float inv = 1.0f / sh[0];
    for (int i = threadIdx.x; i < n; i += 256) out[i] = expf(p[i] - m) * inv;
}

__global__ __launch_bounds__(256) void k_scale_cols(
    const float* __restrict__ in, const float* __restrict__ sm,
    float* __restrict__ out, int total, int C)
{
    for (int i = blockIdx.x * 256 + threadIdx.x; i < total; i += gridDim.x * 256)
        out[i] = in[i] * sm[i % C];
}

__global__ __launch_bounds__(256) void k_mul(
    const float* __restrict__ a, const float* __restrict__ b,
    float* __restrict__ c, int total)
{
    for (int i = blockIdx.x * 256 + threadIdx.x; i < total; i += gridDim.x * 256)
        c[i] = a[i] * b[i];
}

__global__ __launch_bounds__(256) void k_add(
    const float* __restrict__ a, const float* __restrict__ b,
    float* __restrict__ c, int total)
{
    for (int i = blockIdx.x * 256 + threadIdx.x; i < total; i += gridDim.x * 256)
        c[i] = a[i] + b[i];
}

// ---------------------------------------------------------------------------
// GCA edge attention: global-per-head softmax over the edge dimension.
// ---------------------------------------------------------------------------
__device__ __forceinline__ unsigned fenc(float f) {
    unsigned u = __float_as_uint(f);
    return (u & 0x80000000u) ? ~u : (u | 0x80000000u);
}
__device__ __forceinline__ float fdec(unsigned u) {
    unsigned v = (u & 0x80000000u) ? (u & 0x7FFFFFFFu) : ~u;
    return __uint_as_float(v);
}

__global__ __launch_bounds__(256) void k_edge_scores(
    const float* __restrict__ q, const float* __restrict__ k,
    const int* __restrict__ src, const int* __restrict__ dst,
    float* __restrict__ es, int E)
{
    int idx = blockIdx.x * 256 + threadIdx.x;
    if (idx >= E * NHD) return;
    int e = idx >> 3, h = idx & 7;
    const v4f* qp = (const v4f*)(q + (size_t)dst[e] * HH + h * HDD);
    const v4f* kp = (const v4f*)(k + (size_t)src[e] * HH + h * HDD);
    float acc = 0.0f;
    #pragma unroll
    for (int i = 0; i < 8; ++i) {
        v4f a = qp[i], b = kp[i];
        acc += a[0]*b[0] + a[1]*b[1] + a[2]*b[2] + a[3]*b[3];
    }
    es[idx] = acc * 0.17677669529663689f;  // 1/sqrt(32)
}

__global__ void k_init_red(unsigned* redm, float* reds)
{
    if (threadIdx.x < NHD) { redm[threadIdx.x] = fenc(-INFINITY); reds[threadIdx.x] = 0.0f; }
}

__global__ __launch_bounds__(256) void k_red_max(
    const float* __restrict__ es, unsigned* __restrict__ redm, int total)
{
    __shared__ unsigned sm[NHD];
    if (threadIdx.x < NHD) sm[threadIdx.x] = fenc(-INFINITY);
    __syncthreads();
    for (int i = blockIdx.x * 256 + threadIdx.x; i < total; i += gridDim.x * 256)
        atomicMax(&sm[i & 7], fenc(es[i]));
    __syncthreads();
    if (threadIdx.x < NHD) atomicMax(&redm[threadIdx.x], sm[threadIdx.x]);
}

__global__ __launch_bounds__(256) void k_exp_sum(
    float* __restrict__ es, const unsigned* __restrict__ redm,
    float* __restrict__ reds, int total)
{
    __shared__ float ss[NHD];
    __shared__ float mx[NHD];
    if (threadIdx.x < NHD) { ss[threadIdx.x] = 0.0f; mx[threadIdx.x] = fdec(redm[threadIdx.x]); }
    __syncthreads();
    for (int i = blockIdx.x * 256 + threadIdx.x; i < total; i += gridDim.x * 256) {
        float e = expf(es[i] - mx[i & 7]);
        es[i] = e;
        atomicAdd(&ss[i & 7], e);
    }
    __syncthreads();
    if (threadIdx.x < NHD) atomicAdd(&reds[threadIdx.x], ss[threadIdx.x]);
}

__global__ __launch_bounds__(256) void k_scatter(
    const float* __restrict__ es, const float* __restrict__ reds,
    const float* __restrict__ v, const int* __restrict__ src,
    const int* __restrict__ dst, float* __restrict__ agg, int E)
{
    int idx = blockIdx.x * 256 + threadIdx.x;
    if (idx >= E * NHD) return;
    int e = idx >> 3, h = idx & 7;
    float w = es[idx] / reds[h];
    const v4f* vp = (const v4f*)(v + (size_t)src[e] * HH + h * HDD);
    float* ap = agg + (size_t)dst[e] * HH + h * HDD;
    #pragma unroll
    for (int i = 0; i < 8; ++i) {
        v4f vv = vp[i];
        atomicAdd(&ap[i * 4 + 0], vv[0] * w);
        atomicAdd(&ap[i * 4 + 1], vv[1] * w);
        atomicAdd(&ap[i * 4 + 2], vv[2] * w);
        atomicAdd(&ap[i * 4 + 3], vv[3] * w);
    }
}

// ---------------------------------------------------------------------------
// Host-side helpers
// ---------------------------------------------------------------------------
static inline void gemm(hipStream_t s, const float* A, int lda,
                        const float* B, int ldb, int tB,
                        float* C, int ldc, const float* bias,
                        int M, int N, int K, float alpha, int act, int addC)
{
    dim3 grid((N + BN - 1) / BN, (M + BM - 1) / BM);
    k_wmma_gemm<<<grid, dim3(256), 0, s>>>(A, lda, B, ldb, tB, C, ldc, bias,
                                           M, N, K, alpha, act, addC);
}

static inline void lnorm(hipStream_t s, const float* X, int ldx, float* Y, int ldy,
                         const float* g, const float* b, int D, int act)
{
    k_layernorm<<<dim3(NN), dim3(256), 0, s>>>(X, ldx, Y, ldy, g, b, D, act);
}

// ---------------------------------------------------------------------------
// kernel_launch
//
// Input order: d_in[0]=temporal (2048x64), d_in[1]=gene (2048x128),
// d_in[2]=topology (2048x64), d_in[3]=edge_index (2x262144),
// d_in[4..170] = params pytree leaves, flattened with dicts in sorted-key
// order and lists in order (JAX pytree convention):
//   fusion.g extractor [4..21], fusion.gate [22,23], fusion.out_lin [24,25],
//   fusion.out_ln [26,27], fusion.p extractor [28..45], fusion.t [46..63],
//   gca[i] (f1.b,f1.w,f2.b,f2.w,k.b,k.w,ln1.b,ln1.g,ln2.b,ln2.g,o.b,o.w,
//           q.b,q.w,v.b,v.w) at [64+16i .. 79+16i],
//   imp_g=112, imp_p=113, imp_t=114,
//   msa.fusion_lin [115,116], msa.fusion_ln [117,118], msa.out_lin [119,120],
//   msa.out_ln [121,122], msa.scales[s] (trans_lin.b/w, trans_ln.b/g,
//     wk.b/w, wo.b/w, wq.b/w, wv.b/w) at [123+12s .. 134+12s],
//   norms[i] (b,g) at [159+2i,160+2i],
//   pred.l1 [165,166], pred.l2 [167,168], pred.ln [169,170].
// Extractor leaf layout: layers i: lin.b,lin.w,ln.b,ln.g (4 each, i=0..2),
// then skips i: b,w (2 each).
// ---------------------------------------------------------------------------
extern "C" void kernel_launch(void* const* d_in, const int* in_sizes, int n_in,
                              void* d_out, int out_size, void* d_ws, size_t ws_size,
                              hipStream_t stream)
{
    const float* tf   = (const float*)d_in[0];
    const float* gf   = (const float*)d_in[1];
    const float* pfin = (const float*)d_in[2];
    const int*   edge = (const int*)d_in[3];
    const int    E    = in_sizes[3] / 2;
    const int*   src  = edge;
    const int*   dst  = edge + E;

    auto P = [&](int i) -> const float* { return (const float*)d_in[4 + i]; };

    // ---- workspace bump allocator ----
    char* wp = (char*)d_ws;
    auto alloc = [&](size_t nf) -> float* {
        float* p = (float*)wp;
        wp += ((nf * sizeof(float)) + 255) & ~(size_t)255;
        return p;
    };
    float* smv   = alloc(1024);              // softmax(imp_*): t@0, g@256, p@512
    float* tfs   = alloc((size_t)NN * 64);
    float* gfs   = alloc((size_t)NN * 128);
    float* pfs   = alloc((size_t)NN * 64);
    float* tG    = alloc((size_t)NN * HH);   // generic GEMM target
    float* p0    = alloc((size_t)NN * HH);   // extractor ping
    float* p1    = alloc((size_t)NN * HH);   // extractor pong
    float* comb  = alloc((size_t)NN * 768);
    float* gateb = alloc((size_t)NN * 768);
    float* xbuf  = alloc((size_t)NN * HH);   // main residual stream
    float* t1    = alloc((size_t)NN * HH);
    float* t2    = alloc((size_t)NN * HH);
    float* qb    = alloc((size_t)NN * HH);
    float* kb    = alloc((size_t)NN * HH);
    float* vb    = alloc((size_t)NN * HH);
    float* scb   = alloc((size_t)NN * NN);   // per-head attention scores
    float* mtmp  = alloc((size_t)NN * HH);   // pre-Wo MHA output
    float* mcomb = alloc((size_t)NN * 768);  // MSA feats concat
    float* agg   = alloc((size_t)NN * HH);
    float* ffb   = alloc((size_t)NN * 1024);
    float* esc   = alloc((size_t)E * NHD);
    unsigned* redm = (unsigned*)alloc(16);
    float*    reds = (float*)(redm + NHD);
    float* h1    = alloc((size_t)NN * 128);
    float* h1g   = alloc((size_t)NN * 128);

    const int EB = (E * NHD + 255) / 256;

    // ---- 1. importance softmax + column scaling ----
    k_softmax_vec<<<1, 256, 0, stream>>>(P(114), 64,  smv + 0);    // imp_t
    k_softmax_vec<<<1, 256, 0, stream>>>(P(112), 128, smv + 256);  // imp_g
    k_softmax_vec<<<1, 256, 0, stream>>>(P(113), 64,  smv + 512);  // imp_p
    k_scale_cols<<<512, 256, 0, stream>>>(tf,   smv + 0,   tfs, NN * 64,  64);
    k_scale_cols<<<512, 256, 0, stream>>>(gf,   smv + 256, gfs, NN * 128, 128);
    k_scale_cols<<<512, 256, 0, stream>>>(pfin, smv + 512, pfs, NN * 64,  64);

    // ---- 2. extractors -> concat into comb [N,768] ----
    auto extractor = [&](const float* X0, int din, int base, float* outp, int outld) {
        const float* xc = X0; int ck = din;
        float* pp[2] = { p0, p1 };
        for (int i = 0; i < 3; ++i) {
            const float* linb = P(base + i * 4 + 0);
            const float* linw = P(base + i * 4 + 1);
            const float* lnb  = P(base + i * 4 + 2);
            const float* lng  = P(base + i * 4 + 3);
            const float* skb  = P(base + 12 + i * 2 + 0);
            const float* skw  = P(base + 12 + i * 2 + 1);
            gemm(stream, xc, ck, linw, HH, 0, tG, HH, linb, NN, HH, ck, 1.f, 0, 0);
            float* nxt = (i == 2) ? outp : pp[i & 1];
            int    nld = (i == 2) ? outld : HH;
            lnorm(stream, tG, HH, nxt, nld, lng, lnb, HH, 1);              // relu(LN)
            gemm(stream, xc, ck, skw, HH, 0, nxt, nld, skb, NN, HH, ck, 1.f, 0, 1); // += skip
            xc = nxt; ck = HH;
        }
    };
    extractor(tfs, 64,  46, comb + 0,   768);  // fusion.t
    extractor(gfs, 128, 4,  comb + 256, 768);  // fusion.g
    extractor(pfs, 64,  28, comb + 512, 768);  // fusion.p

    // ---- 3. gated fusion ----
    gemm(stream, comb, 768, P(23), 768, 0, gateb, 768, P(22), NN, 768, 768, 1.f, 2, 0);
    k_mul<<<2048, 256, 0, stream>>>(comb, gateb, comb, NN * 768);          // fused
    gemm(stream, comb, 768, P(25), HH, 0, tG, HH, P(24), NN, HH, 768, 1.f, 0, 0);
    lnorm(stream, tG, HH, xbuf, HH, P(27), P(26), HH, 1);                   // x

    // ---- 4. multi-scale attention ----
    for (int s = 0; s < 3; ++s) {
        const int b = 123 + 12 * s;
        // sx = relu(LN(trans(x)))
        gemm(stream, xbuf, HH, P(b + 1), HH, 0, tG, HH, P(b + 0), NN, HH, HH, 1.f, 0, 0);
        lnorm(stream, tG, HH, t2, HH, P(b + 3), P(b + 2), HH, 1);
        // q, k, v
        gemm(stream, t2, HH, P(b + 9),  HH, 0, qb, HH, P(b + 8),  NN, HH, HH, 1.f, 0, 0);
        gemm(stream, t2, HH, P(b + 5),  HH, 0, kb, HH, P(b + 4),  NN, HH, HH, 1.f, 0, 0);
        gemm(stream, t2, HH, P(b + 11), HH, 0, vb, HH, P(b + 10), NN, HH, HH, 1.f, 0, 0);
        for (int h = 0; h < NHD; ++h) {
            // scores = (Qh @ Kh^T) / sqrt(HD)   (transB path: N=2048, ldb=256)
            gemm(stream, qb + h * HDD, HH, kb + h * HDD, HH, 1, scb, NN,
                 nullptr, NN, NN, HDD, 0.17677669529663689f, 0, 0);
            k_row_softmax<<<dim3(NN), dim3(256), 0, stream>>>(scb, NN);
            // out_h = attn @ Vh
            gemm(stream, scb, NN, vb + h * HDD, HH, 0, mtmp + h * HDD, HH,
                 nullptr, NN, HDD, NN, 1.f, 0, 0);
        }
        // feats[s] = Wo(mtmp) -> mcomb columns
        gemm(stream, mtmp, HH, P(b + 7), HH, 0, mcomb + s * HH, 768, P(b + 6),
             NN, HH, HH, 1.f, 0, 0);
    }
    // fused = relu(LN(fusion_lin(mcomb)))
    gemm(stream, mcomb, 768, P(116), HH, 0, tG, HH, P(115), NN, HH, 768, 1.f, 0, 0);
    lnorm(stream, tG, HH, t2, HH, P(118), P(117), HH, 1);
    k_add<<<512, 256, 0, stream>>>(t2, xbuf, t1, NN * HH);                 // fused + x
    gemm(stream, t1, HH, P(120), HH, 0, tG, HH, P(119), NN, HH, HH, 1.f, 0, 0);
    lnorm(stream, tG, HH, xbuf, HH, P(122), P(121), HH, 1);                // new x

    // ---- 5. graph cross-attention x3 ----
    for (int i = 0; i < 3; ++i) {
        const int b = 64 + 16 * i;
        lnorm(stream, xbuf, HH, t1, HH, P(b + 7), P(b + 6), HH, 0);        // h = LN1(x)
        gemm(stream, t1, HH, P(b + 13), HH, 0, qb, HH, P(b + 12), NN, HH, HH, 1.f, 0, 0);
        gemm(stream, t1, HH, P(b + 5),  HH, 0, kb, HH, P(b + 4),  NN, HH, HH, 1.f, 0, 0);
        gemm(stream, t1, HH, P(b + 15), HH, 0, vb, HH, P(b + 14), NN, HH, HH, 1.f, 0, 0);
        // edge scores + global per-head softmax + weighted scatter
        k_edge_scores<<<EB, 256, 0, stream>>>(qb, kb, src, dst, esc, E);
        k_init_red<<<1, 32, 0, stream>>>(redm, reds);
        k_red_max<<<4096, 256, 0, stream>>>(esc, redm, E * NHD);
        k_exp_sum<<<4096, 256, 0, stream>>>(esc, redm, reds, E * NHD);
        hipMemsetAsync(agg, 0, (size_t)NN * HH * sizeof(float), stream);
        k_scatter<<<EB, 256, 0, stream>>>(esc, reds, vb, src, dst, agg, E);
        // out proj + residual
        gemm(stream, agg, HH, P(b + 11), HH, 0, t2, HH, P(b + 10), NN, HH, HH, 1.f, 0, 0);
        k_add<<<512, 256, 0, stream>>>(xbuf, t2, t1, NN * HH);             // x1
        // FFN
        lnorm(stream, t1, HH, t2, HH, P(b + 9), P(b + 8), HH, 0);
        gemm(stream, t2, HH, P(b + 1), 1024, 0, ffb, 1024, P(b + 0), NN, 1024, HH, 1.f, 1, 0);
        gemm(stream, ffb, 1024, P(b + 3), HH, 0, t2, HH, P(b + 2), NN, HH, 1024, 1.f, 0, 0);
        k_add<<<512, 256, 0, stream>>>(t1, t2, t2, NN * HH);               // gca_out
        // x = LN(norms[i], x + gca_out)
        k_add<<<512, 256, 0, stream>>>(xbuf, t2, t1, NN * HH);
        lnorm(stream, t1, HH, xbuf, HH, P(160 + 2 * i), P(159 + 2 * i), HH, 0);
    }

    // ---- 6. prediction head ----
    gemm(stream, xbuf, HH, P(166), 128, 0, h1, 128, P(165), NN, 128, HH, 1.f, 0, 0);
    lnorm(stream, h1, 128, h1g, 128, P(170), P(169), 128, 2);              // gelu(LN)
    gemm(stream, h1g, 128, P(168), 1, 0, (float*)d_out, 1, P(167), NN, 1, 128, 1.f, 0, 0);

    (void)n_in; (void)out_size; (void)ws_size;
}
